// MultiHeadAttention_26482768347412
// MI455X (gfx1250) — matrix operand order repaired and checked by reading, not compiled, not run on hardware
//
#include <hip/hip_runtime.h>
#include <hip/hip_bf16.h>

// ---------------------------------------------------------------------------
// MHA block for MI455X (gfx1250, wave32, WMMA bf16 16x16x32)
// B=4, S=1024, D=1024, H=16, DK=DV=64
// Fused scores->softmax->ctx kernel staging the score slab in WGP LDS,
// with K tiles staged via async global->LDS (ASYNCcnt) shared across waves.
// ---------------------------------------------------------------------------

#define Bsz 4
#define Ssz 1024
#define Dsz 1024
#define Hsz 16
#define DKsz 64
#define PLD 1028        // LDS score-row stride (floats): 16B aligned, bank-friendly
#define KLD 72          // LDS K-stage row stride (bf16): 144B, bank-friendly

typedef __attribute__((ext_vector_type(16))) __bf16 v16bf;
typedef __attribute__((ext_vector_type(8)))  __bf16 v8bf;
typedef __attribute__((ext_vector_type(8)))  float  v8f;
typedef __attribute__((ext_vector_type(4)))  float  v4f;

// ---- WMMA fragment loaders (A row / B col = lane%16, contiguous k = 16*(lane/16) + e, wave32) ----------
__device__ __forceinline__ v16bf load_a_bf16(const __bf16* base, int ldk, int lane) {
    const __bf16* row = base + (size_t)(lane & 15) * ldk + ((lane >> 4) & 1) * 16;
    v8bf lo = *(const v8bf*)(row);
    v8bf hi = *(const v8bf*)(row + 8);
    v16bf a;
#pragma unroll
    for (int e = 0; e < 8; ++e) { a[e] = lo[e]; a[e + 8] = hi[e]; }
    return a;
}

__device__ __forceinline__ v16bf load_a_f32(const float* base, int ldk, int lane) {
    const float* row = base + (size_t)(lane & 15) * ldk + ((lane >> 4) & 1) * 16;
    v16bf a;
#pragma unroll
    for (int e = 0; e < 8; ++e)  a[e] = (__bf16)row[e];
#pragma unroll
    for (int e = 8; e < 16; ++e) a[e] = (__bf16)row[e];
    return a;
}

__device__ __forceinline__ v16bf load_b_bf16(const __bf16* base, int ldk, int lane) {
    const __bf16* col = base + (size_t)(lane & 15) * ldk + ((lane >> 4) & 1) * 16;
    return *(const v16bf*)col;
}

// B fragment from a 16B-aligned LDS stage (two 16B reads).
__device__ __forceinline__ v16bf load_b_lds(const __bf16* base, int ldk, int lane) {
    const __bf16* col = base + (lane & 15) * ldk + ((lane >> 4) & 1) * 16;
    v8bf lo = *(const v8bf*)(col);
    v8bf hi = *(const v8bf*)(col + 8);
    v16bf r;
#pragma unroll
    for (int e = 0; e < 8; ++e) { r[e] = lo[e]; r[e + 8] = hi[e]; }
    return r;
}

__device__ __forceinline__ v8f wmma_bf16(v16bf a, v16bf b, v8f c) {
    return __builtin_amdgcn_wmma_f32_16x16x32_bf16(false, a, false, b, (short)0, c,
                                                   false, false);
}

__device__ __forceinline__ v8f vzero8() { return (v8f){0.f,0.f,0.f,0.f,0.f,0.f,0.f,0.f}; }

// Async global->LDS copy of 16B per lane (CDNA5 GLOBAL_LOAD_ASYNC_TO_LDS_B128,
// tracked by ASYNCcnt).
__device__ __forceinline__ void async_g2l_b128(uint32_t ldsOff, const void* src) {
    asm volatile("global_load_async_to_lds_b128 %0, %1, off"
                 :: "v"(ldsOff), "v"((uint64_t)(uintptr_t)src)
                 : "memory");
}
__device__ __forceinline__ void wait_asynccnt0() {
    asm volatile("s_wait_asynccnt 0" ::: "memory");
}

// ---------------------------------------------------------------------------
// Elementwise cast f32 -> bf16, 4 elements/thread.
// ---------------------------------------------------------------------------
__global__ __launch_bounds__(256) void cast_kernel(const float* __restrict__ X,
                                                   __bf16* __restrict__ Y) {
    const size_t i = ((size_t)blockIdx.x * 256 + threadIdx.x) * 4;
    v4f x = *(const v4f*)(X + i);
    __bf16 y[4];
#pragma unroll
    for (int e = 0; e < 4; ++e) y[e] = (__bf16)x[e];
    *(uint64_t*)(Y + i) = *(const uint64_t*)y;
}

// ---------------------------------------------------------------------------
// Mask -> additive f32 bias: mb[i] = mask[i] ? -1e9 : 0.
// ---------------------------------------------------------------------------
__global__ __launch_bounds__(256) void maskbias_kernel(const unsigned char* __restrict__ mask,
                                                       float* __restrict__ mb) {
    const size_t i = ((size_t)blockIdx.x * 256 + threadIdx.x) * 4;
    uint32_t m4 = *(const uint32_t*)(mask + i);
    v4f o;
#pragma unroll
    for (int e = 0; e < 4; ++e) o[e] = ((m4 >> (8 * e)) & 0xff) ? -1e9f : 0.f;
    *(v4f*)(mb + i) = o;
}

// ---------------------------------------------------------------------------
// Weight transpose + cast via LDS: W[K][N] f32 -> Wt[N][K] bf16 (1024x1024).
// ---------------------------------------------------------------------------
__global__ __launch_bounds__(256) void wtrans_kernel(const float* __restrict__ W,
                                                     __bf16* __restrict__ Wt) {
    __shared__ __bf16 tile[32][33];
    const int tx = threadIdx.x & 31;
    const int ty = threadIdx.x >> 5;
    const int k0 = blockIdx.x * 32;
    const int n0 = blockIdx.y * 32;
#pragma unroll
    for (int i = 0; i < 32; i += 8)
        tile[ty + i][tx] = (__bf16)W[(size_t)(k0 + ty + i) * 1024 + n0 + tx];
    __syncthreads();
#pragma unroll
    for (int i = 0; i < 32; i += 8)
        Wt[(size_t)(n0 + ty + i) * 1024 + k0 + tx] = tile[tx][ty + i];
}

// ---------------------------------------------------------------------------
// GEMM: Y[4096][1024] = (A bf16 @ Wt bf16 + bias) * scale
// MODE 0: bf16 natural; MODE 1: bf16 head-transposed v_t[b][h][d][s]; MODE 2: f32.
// ---------------------------------------------------------------------------
template <int MODE>
__global__ __launch_bounds__(256) void gemm_kernel(const __bf16* __restrict__ A,
                                                   const __bf16* __restrict__ Wt,
                                                   const float* __restrict__ bias,
                                                   __bf16* __restrict__ outb,
                                                   float* __restrict__ outf,
                                                   float scale) {
    const int lane = threadIdx.x & 31;
    const int wave = threadIdx.x >> 5;
    const int rowBase = blockIdx.x * 256 + wave * 32;
    const int colBase = blockIdx.y * 64;

    const __bf16* Arow0 = A + (size_t)rowBase * 1024;
    const __bf16* Arow1 = A + (size_t)(rowBase + 16) * 1024;

    v8f acc[2][4];
#pragma unroll
    for (int i = 0; i < 2; ++i)
#pragma unroll
        for (int j = 0; j < 4; ++j) acc[i][j] = vzero8();

    for (int kt = 0; kt < 32; ++kt) {
        const int kw = kt * 32;
        if (kt + 4 < 32) {                        // gfx1250 global_prefetch_b8
            __builtin_prefetch(Arow0 + kw + 128 + (size_t)(lane & 15) * 1024, 0, 1);
            __builtin_prefetch(Wt + (size_t)(colBase + (lane & 15)) * 1024 + kw + 128, 0, 1);
        }
        v16bf a0 = load_a_bf16(Arow0 + kw, 1024, lane);
        v16bf a1 = load_a_bf16(Arow1 + kw, 1024, lane);
        v16bf bfrag[4];
#pragma unroll
        for (int j = 0; j < 4; ++j)
            bfrag[j] = load_b_bf16(Wt + (size_t)(colBase + j * 16) * 1024 + kw, 1024, lane);
#pragma unroll
        for (int j = 0; j < 4; ++j) {
            acc[0][j] = wmma_bf16(a0, bfrag[j], acc[0][j]);
            acc[1][j] = wmma_bf16(a1, bfrag[j], acc[1][j]);
        }
    }

    const int nloc = lane & 15;
    const int moff = (lane >> 4) * 8;
#pragma unroll
    for (int i = 0; i < 2; ++i) {
#pragma unroll
        for (int j = 0; j < 4; ++j) {
            const int n = colBase + j * 16 + nloc;
            const float bv = bias[n];
#pragma unroll
            for (int r = 0; r < 8; ++r) {
                const int m = rowBase + i * 16 + moff + r;
                const float val = (acc[i][j][r] + bv) * scale;
                if constexpr (MODE == 0) {
                    outb[(size_t)m * 1024 + n] = (__bf16)val;
                } else if constexpr (MODE == 1) {
                    const int b = m >> 10, s = m & 1023;
                    const int h = n >> 6,  d = n & 63;
                    outb[(((size_t)(b * Hsz + h) * DKsz) + d) * Ssz + s] = (__bf16)val;
                } else {
                    outf[(size_t)m * 1024 + n] = val;
                }
            }
        }
    }
}

// ---------------------------------------------------------------------------
// Fused attention: K tile async-staged to LDS (shared by all 4 waves),
// scores (QK^T + maskbias) -> LDS slab -> softmax -> attn out (b128) ->
// ctx = P @ V (WMMA, A from LDS).
// Block 128 (4 waves); wave owns 16 q-rows; WG slab = 64 x PLD f32 + K stage.
// grid (B*H, S/64). q pre-scaled by 0.125 at projection.
// ---------------------------------------------------------------------------
__global__ __launch_bounds__(128) void fusedattn_kernel(const __bf16* __restrict__ qb,
                                                        const __bf16* __restrict__ kb,
                                                        const __bf16* __restrict__ vt,
                                                        const float* __restrict__ mbias,
                                                        float* __restrict__ attn,
                                                        __bf16* __restrict__ ctx) {
    extern __shared__ float Psh[];
    const int tid  = threadIdx.x;
    const int lane = tid & 31;
    const int wave = tid >> 5;                  // 0..3
    const int bh = blockIdx.x;
    const int b = bh >> 4, h = bh & 15;
    const int qrow0 = blockIdx.y * 64 + wave * 16;
    float*  P      = Psh + (size_t)wave * 16 * PLD;
    __bf16* Kstage = (__bf16*)(Psh + (size_t)64 * PLD);      // [16][KLD] bf16
    const uint32_t kstageOff = (uint32_t)(uintptr_t)Kstage;  // LDS byte offset

    const int nloc = lane & 15;
    const int moff = (lane >> 4) * 8;

    // Per-thread slice of the K-tile copy: 128 threads x 16B = 2KB tile.
    const int crow = tid >> 3;                  // 0..15
    const int cchk = tid & 7;                   // 0..7  (16B chunks of 128B row)
    const uint32_t ldsDst = kstageOff + (uint32_t)(crow * KLD * 2 + cchk * 16);
    const __bf16* ksrc0 = kb + ((size_t)b * Ssz + crow) * 1024 + h * 64 + cchk * 8;

    // ---- Phase 1: scores tiles -> LDS -------------------------------------
    const __bf16* qbase = qb + ((size_t)b * Ssz + qrow0) * 1024 + h * 64;
    v16bf a0 = load_a_bf16(qbase, 1024, lane);
    v16bf a1 = load_a_bf16(qbase + 32, 1024, lane);

    const float* mb = mbias + ((size_t)b * Ssz + qrow0 + moff) * Ssz + nloc;
    float* pdst = P + (size_t)moff * PLD + nloc;

    for (int kblk = 0; kblk < 64; ++kblk) {
        // async-stage K tile (16 rows x 64 dims bf16) once per WG
        async_g2l_b128(ldsDst, ksrc0 + (size_t)kblk * 16 * 1024);
        wait_asynccnt0();
        __syncthreads();

        v16bf b0 = load_b_lds(Kstage, KLD, lane);
        v16bf b1 = load_b_lds(Kstage + 32, KLD, lane);
        const int n = kblk * 16;
        v8f c;
#pragma unroll
        for (int r = 0; r < 8; ++r) c[r] = mb[(size_t)r * Ssz + n];
        c = wmma_bf16(a0, b0, c);
        c = wmma_bf16(a1, b1, c);
#pragma unroll
        for (int r = 0; r < 8; ++r) pdst[(size_t)r * PLD + n] = c[r];
        __syncthreads();                       // all waves done with stage
    }

    // ---- Phase 2: row softmax out of LDS; write attn once (b128) ----------
    float* arow_base = attn + ((size_t)bh * Ssz + qrow0) * Ssz;
    for (int m = 0; m < 16; ++m) {
        float* prow = P + (size_t)m * PLD;
        v4f t[8];
        float mx = -3.4e38f;
#pragma unroll
        for (int i = 0; i < 8; ++i) {
            t[i] = *(v4f*)(prow + lane * 4 + 128 * i);
#pragma unroll
            for (int j = 0; j < 4; ++j) mx = fmaxf(mx, t[i][j]);
        }
#pragma unroll
        for (int off = 16; off > 0; off >>= 1) mx = fmaxf(mx, __shfl_xor(mx, off, 32));

        float sum = 0.f;
#pragma unroll
        for (int i = 0; i < 8; ++i)
#pragma unroll
            for (int j = 0; j < 4; ++j) { t[i][j] = __expf(t[i][j] - mx); sum += t[i][j]; }
#pragma unroll
        for (int off = 16; off > 0; off >>= 1) sum += __shfl_xor(sum, off, 32);

        const float inv = 1.f / sum;
        float* ar = arow_base + (size_t)m * Ssz;
#pragma unroll
        for (int i = 0; i < 8; ++i) {
            v4f o = t[i] * inv;
            *(v4f*)(prow + lane * 4 + 128 * i) = o;       // LDS (for phase 3)
            *(v4f*)(ar + lane * 4 + 128 * i) = o;         // global attn, coalesced
        }
    }

    // ---- Phase 3: ctx = P @ V (A frags from LDS, cvt f32->bf16) -----------
    const __bf16* vhead = vt + (size_t)bh * DKsz * Ssz;
    v8f acc[4];
#pragma unroll
    for (int j = 0; j < 4; ++j) acc[j] = vzero8();

    for (int kt = 0; kt < 32; ++kt) {
        const int kw = kt * 32;
        v16bf a = load_a_f32(P + kw, PLD, lane);
#pragma unroll
        for (int j = 0; j < 4; ++j) {
            v16bf bfr = load_b_bf16(vhead + (size_t)(j * 16) * Ssz + kw, Ssz, lane);
            acc[j] = wmma_bf16(a, bfr, acc[j]);
        }
    }

#pragma unroll
    for (int j = 0; j < 4; ++j) {
        const int d = j * 16 + nloc;
#pragma unroll
        for (int r = 0; r < 8; ++r) {
            const int s = qrow0 + moff + r;
            ctx[((size_t)b * Ssz + s) * 1024 + h * 64 + d] = (__bf16)acc[j][r];
        }
    }
}

// ---------------------------------------------------------------------------
// LayerNorm(out_proj + residual) * g + b -> d_out.
// ---------------------------------------------------------------------------
__global__ __launch_bounds__(256) void ln_kernel(const float* __restrict__ oproj,
                                                 const float* __restrict__ resid,
                                                 const float* __restrict__ g,
                                                 const float* __restrict__ bta,
                                                 float* __restrict__ out) {
    __shared__ float swred[8];
    const int tid = threadIdx.x;
    const int lane = tid & 31;
    const int wid = tid >> 5;
    const size_t row = blockIdx.x;
    const float* xr = oproj + row * 1024;
    const float* rr = resid + row * 1024;

    float x[4];
    float s = 0.f;
#pragma unroll
    for (int i = 0; i < 4; ++i) { x[i] = xr[tid + 256 * i] + rr[tid + 256 * i]; s += x[i]; }
#pragma unroll
    for (int off = 16; off > 0; off >>= 1) s += __shfl_xor(s, off, 32);
    if (lane == 0) swred[wid] = s;
    __syncthreads();
    float tot = 0.f;
#pragma unroll
    for (int w = 0; w < 8; ++w) tot += swred[w];
    const float mu = tot * (1.f / 1024.f);
    __syncthreads();

    float vv = 0.f;
#pragma unroll
    for (int i = 0; i < 4; ++i) { const float d = x[i] - mu; vv += d * d; }
#pragma unroll
    for (int off = 16; off > 0; off >>= 1) vv += __shfl_xor(vv, off, 32);
    if (lane == 0) swred[wid] = vv;
    __syncthreads();
    float vtot = 0.f;
#pragma unroll
    for (int w = 0; w < 8; ++w) vtot += swred[w];
    const float inv = rsqrtf(vtot * (1.f / 1024.f) + 1e-5f);

#pragma unroll
    for (int i = 0; i < 4; ++i) {
        const int c = tid + 256 * i;
        out[row * 1024 + c] = (x[i] - mu) * inv * g[c] + bta[c];
    }
}

// ---------------------------------------------------------------------------
extern "C" void kernel_launch(void* const* d_in, const int* in_sizes, int n_in,
                              void* d_out, int out_size, void* d_ws, size_t ws_size,
                              hipStream_t stream) {
    const float* Q    = (const float*)d_in[0];
    const float* K    = (const float*)d_in[1];
    const float* V    = (const float*)d_in[2];
    const unsigned char* mask = (const unsigned char*)d_in[3];
    const float* Wq   = (const float*)d_in[4];
    const float* bq   = (const float*)d_in[5];
    const float* Wk   = (const float*)d_in[6];
    const float* bk   = (const float*)d_in[7];
    const float* Wv   = (const float*)d_in[8];
    const float* bv   = (const float*)d_in[9];
    const float* Wout = (const float*)d_in[10];
    const float* bout = (const float*)d_in[11];
    const float* ln_g = (const float*)d_in[12];
    const float* ln_b = (const float*)d_in[13];

    float* out  = (float*)d_out;                               // [B,S,D] = 4M f32
    float* attn = (float*)d_out + (size_t)Bsz * Ssz * Dsz;     // [B,H,S,S] = 64M f32

    // Workspace layout
    char* ws = (char*)d_ws;
    const size_t MB = 1024ull * 1024ull;
    __bf16* Wt_q  = (__bf16*)(ws + 0 * MB);
    __bf16* Wt_k  = (__bf16*)(ws + 2 * MB);
    __bf16* Wt_v  = (__bf16*)(ws + 4 * MB);
    __bf16* Wt_o  = (__bf16*)(ws + 6 * MB);
    __bf16* qb    = (__bf16*)(ws + 8 * MB);
    __bf16* kbuf  = (__bf16*)(ws + 16 * MB);
    __bf16* vt    = (__bf16*)(ws + 24 * MB);
    __bf16* Qc    = (__bf16*)(ws + 32 * MB);
    __bf16* Kc    = (__bf16*)(ws + 40 * MB);
    __bf16* Vc    = (__bf16*)(ws + 48 * MB);
    __bf16* ctxb  = (__bf16*)(ws + 56 * MB);
    float*  oprj  = (float*)(ws + 64 * MB);   // 16MB
    float*  mbias = (float*)(ws + 80 * MB);   // [B,S,S] f32 = 16MB

    const int NELT = Bsz * Ssz * Dsz;         // 4M

    // 1) casts, mask bias, weight transposes
    cast_kernel<<<NELT / 1024, 256, 0, stream>>>(Q, Qc);
    cast_kernel<<<NELT / 1024, 256, 0, stream>>>(K, Kc);
    cast_kernel<<<NELT / 1024, 256, 0, stream>>>(V, Vc);
    maskbias_kernel<<<(Bsz * Ssz * Ssz) / 1024, 256, 0, stream>>>(mask, mbias);
    dim3 tgrid(32, 32);
    wtrans_kernel<<<tgrid, 256, 0, stream>>>(Wq, Wt_q);
    wtrans_kernel<<<tgrid, 256, 0, stream>>>(Wk, Wt_k);
    wtrans_kernel<<<tgrid, 256, 0, stream>>>(Wv, Wt_v);
    wtrans_kernel<<<tgrid, 256, 0, stream>>>(Wout, Wt_o);

    // 2) QKV projections (Q pre-scaled by 1/sqrt(dk))
    dim3 ggrid(16, 16);
    gemm_kernel<0><<<ggrid, 256, 0, stream>>>(Qc, Wt_q, bq, qb,   nullptr, 0.125f);
    gemm_kernel<0><<<ggrid, 256, 0, stream>>>(Kc, Wt_k, bk, kbuf, nullptr, 1.0f);
    gemm_kernel<1><<<ggrid, 256, 0, stream>>>(Vc, Wt_v, bv, vt,   nullptr, 1.0f);

    // 3) fused scores + softmax + ctx
    dim3 agrid(Bsz * Hsz, Ssz / 64);
    const size_t ldsBytes = (size_t)64 * PLD * sizeof(float)   // score slab
                          + (size_t)16 * KLD * sizeof(__bf16); // K stage
    fusedattn_kernel<<<agrid, 128, ldsBytes, stream>>>(qb, kbuf, vt, mbias, attn, ctxb);

    // 4) output projection (f32 out)
    gemm_kernel<2><<<ggrid, 256, 0, stream>>>(ctxb, Wt_o, bout, nullptr, oprj, 1.0f);

    // 5) residual + layernorm -> out
    ln_kernel<<<Bsz * Ssz, 256, 0, stream>>>(oprj, Q, ln_g, ln_b, out);
}
